// MultiHeadSelfAttention_49426483642369
// MI455X (gfx1250) — compile-verified, matrix operand order checked by reading, not run on hardware
//
#include <hip/hip_runtime.h>

// ---------------------------------------------------------------------------
// MHSA for MI455X (gfx1250, wave32).
//  - bf16 WMMA (v_wmma_f32_16x16x32_bf16), fp32 accumulate, fp32 softmax
//  - staging via CDNA5 async paths: global_load_async_to_lds_b128 (ASYNCcnt)
//    and the Tensor Data Mover (tensor_load_to_lds, TENSORcnt)
// Roofline: mandatory 537MB fp32 attn write ~23us @ 23.3TB/s is the floor;
// all GEMM compute (69 GFLOP) hides under it on the bf16 matrix pipe.
// ---------------------------------------------------------------------------

typedef __attribute__((ext_vector_type(16))) __bf16 v16bf;
typedef __attribute__((ext_vector_type(8)))  __bf16 v8bf;
typedef __attribute__((ext_vector_type(4)))  __bf16 v4bf;
typedef __attribute__((ext_vector_type(8)))  float  v8f;
typedef __attribute__((ext_vector_type(4)))  unsigned int v4u;
typedef __attribute__((ext_vector_type(8)))  int    v8i;
typedef __attribute__((ext_vector_type(4)))  int    v4i;

#define WMMA_BF16(a, b, c) \
  __builtin_amdgcn_wmma_f32_16x16x32_bf16(false, (a), false, (b), (short)0, (c), false, false)

static __device__ __forceinline__ v16bf cat8(v8bf lo, v8bf hi) {
  return __builtin_shufflevector(lo, hi, 0,1,2,3,4,5,6,7,8,9,10,11,12,13,14,15);
}

// Async global->LDS copy of 16B per active lane (ISA §15.18 op 98, ASYNCcnt).
static __device__ __forceinline__ void async_b128(const void* gp, void* lp) {
  unsigned long long ga = (unsigned long long)(uintptr_t)gp;
  unsigned int       la = (unsigned int)(uintptr_t)lp;   // LDS aperture: low 32b = byte offset
  asm volatile("global_load_async_to_lds_b128 %0, %1, off"
               :: "v"(la), "v"(ga) : "memory");
}
static __device__ __forceinline__ void wait_async0() {
  asm volatile("s_wait_asynccnt 0x0" ::: "memory");
}

// TDM: 2D tile load (tile_d0 elems/row x tile_d1 rows, 2-byte elems, row
// stride `stride` elems) from global `gp` into LDS `lp`. D# packing per ISA
// §8.3/§8.4: group0 = {count=1 | lds_addr | global_addr | type=2},
// group1 = {data_size=1(2B), tensor dims, tile dims, dim0 stride}.
// 6-arg builtin form (clang-23 / therock-10.0 headers).
static __device__ __forceinline__ void tdm_load_2d_bf16(
    const void* gp, void* lp, unsigned tile_d0, unsigned tile_d1,
    unsigned tensor_d0, unsigned tensor_d1, unsigned stride) {
  unsigned long long ga = (unsigned long long)(uintptr_t)gp;
  unsigned int       la = (unsigned int)(uintptr_t)lp;
  v4u g0;
  g0[0] = 1u;                                        // count=1, user descriptor
  g0[1] = la;                                        // lds_addr [63:32]
  g0[2] = (unsigned int)(ga & 0xffffffffu);          // global_addr lo
  g0[3] = (unsigned int)((ga >> 32) & 0x01ffffffu)   // global_addr hi (57-bit)
          | 0x80000000u;                             // type=2 ("image") [127:126]
  v8i g1;
  g1[0] = (int)(1u << 16);                           // data_size=1 -> 2 bytes
  g1[1] = (int)((tensor_d0 & 0xffffu) << 16);        // tensor_dim0[15:0] @ bits 63:48
  g1[2] = (int)(((tensor_d0 >> 16) & 0xffffu)        // tensor_dim0[31:16]
          | ((tensor_d1 & 0xffffu) << 16));          // tensor_dim1[15:0]
  g1[3] = (int)(((tensor_d1 >> 16) & 0xffffu)        // tensor_dim1[31:16]
          | ((tile_d0 & 0xffffu) << 16));            // tile_dim0 @ bits 127:112
  g1[4] = (int)(tile_d1 & 0xffffu);                  // tile_dim1 @ bits 143:128
  g1[5] = (int)stride;                               // tensor_dim0_stride[31:0]
  g1[6] = 0;                                         // stride hi | dim1_stride lo
  g1[7] = 0;
  v4i z4 = {};                                       // groups 2/3: NULL (<=2D)
  v8i z8 = {};
  __builtin_amdgcn_tensor_load_to_lds(g0, g1, z4, z4, z8, 0);
}

// ---------------------------------------------------------------------------
// K0: fp32 -> bf16 convert, 4 elems/thread
// ---------------------------------------------------------------------------
__global__ void k_cvt_bf16(const float* __restrict__ in, __bf16* __restrict__ out, int n4) {
  int i = blockIdx.x * blockDim.x + threadIdx.x;
  if (i >= n4) return;
  float4 f = ((const float4*)in)[i];
  v4bf o;
  o[0] = (__bf16)f.x; o[1] = (__bf16)f.y; o[2] = (__bf16)f.z; o[3] = (__bf16)f.w;
  ((v4bf*)out)[i] = o;
}

// ---------------------------------------------------------------------------
// K1/K3: Y = X @ W^T + bias.  X:[4096,1024]bf16, W:[1024,1024]bf16.
// Block = 128 thr (4 waves). Block tile 32(M) x 64(N); per wave: one B frag
// (16 N cols), two A frags (M halves) -> 2 WMMAs/K-step, 2 accumulators.
// A tile staged with async-to-LDS; W tile staged with TDM (wave 0 issues).
// mode 0: bf16 scatter into [b,h,s,dk]; mode 1: fp32 row-major.
// ---------------------------------------------------------------------------
__global__ __launch_bounds__(128) void k_proj(
    const __bf16* __restrict__ X,
    const __bf16* __restrict__ W0, const __bf16* __restrict__ W1, const __bf16* __restrict__ W2,
    const float*  __restrict__ B0, const float*  __restrict__ B1, const float*  __restrict__ B2,
    __bf16* __restrict__ O0, __bf16* __restrict__ O1, __bf16* __restrict__ O2,
    float* __restrict__ OF, int mode)
{
  __shared__ __align__(16) __bf16 At[32 * 32];   // 2 KB
  __shared__ __align__(16) __bf16 Bt[64 * 32];   // 4 KB

  const int tid  = threadIdx.x;
  const int lane = tid & 31;
  const int wave = tid >> 5;
  const int m0   = blockIdx.x * 32;
  const int n0   = blockIdx.y * 64;
  const int z    = blockIdx.z;

  const __bf16* W  = (z == 0) ? W0 : ((z == 1) ? W1 : W2);
  const float*  Bi = (z == 0) ? B0 : ((z == 1) ? B1 : B2);
  __bf16*       O  = (z == 0) ? O0 : ((z == 1) ? O1 : O2);

  const int fm = lane & 15;          // A row / B col within fragment
  const int kb = (lane >> 4) * 8;    // half-lane K split (ISA 7.12.2)

  const int arow = tid >> 2, acol = tid & 3;     // A tile: 128 x 16B

  v8f acc0 = {}, acc1 = {};
  for (int k0 = 0; k0 < 1024; k0 += 32) {
    __syncthreads();
    // A tile 32x32 bf16: one async b128 per thread
    async_b128(&X[(size_t)(m0 + arow) * 1024 + k0 + acol * 8],
               &At[arow * 32 + acol * 8]);
    // B tile 64x32 bf16 via Tensor Data Mover (one descriptor, wave 0)
    if (wave == 0) {
      tdm_load_2d_bf16(&W[(size_t)n0 * 1024 + k0], &Bt[0],
                       /*tile*/ 32, 64, /*tensor*/ 1024, 1024, /*stride*/ 1024);
      __builtin_amdgcn_s_wait_tensorcnt(0);
    }
    wait_async0();
    __syncthreads();

    v16bf a0 = cat8(*(const v8bf*)&At[fm * 32 + kb],
                    *(const v8bf*)&At[fm * 32 + 16 + kb]);
    v16bf a1 = cat8(*(const v8bf*)&At[(16 + fm) * 32 + kb],
                    *(const v8bf*)&At[(16 + fm) * 32 + 16 + kb]);
    int br = wave * 16 + fm;
    v16bf b  = cat8(*(const v8bf*)&Bt[br * 32 + kb],
                    *(const v8bf*)&Bt[br * 32 + 16 + kb]);
    acc0 = WMMA_BF16(a0, b, acc0);
    acc1 = WMMA_BF16(a1, b, acc1);
  }

  // epilogue: C/D layout — lanes 0-15: M=v, lanes 16-31: M=v+8, N=lane&15
  const int e = n0 + wave * 16 + fm;
  const float bias = Bi[e];
#pragma unroll
  for (int v = 0; v < 8; ++v) {
    int mm = v + 8 * (lane >> 4);
#pragma unroll
    for (int half = 0; half < 2; ++half) {
      int   sg  = m0 + half * 16 + mm;                 // b*2048 + s
      float val = (half ? acc1[v] : acc0[v]) + bias;
      if (mode == 0) {
        int bb = sg >> 11, s = sg & 2047, h = e >> 6, dk = e & 63;
        O[((size_t)(bb * 16 + h) * 2048 + s) * 64 + dk] = (__bf16)val;
      } else {
        OF[(size_t)sg * 1024 + e] = val;
      }
    }
  }
}

// ---------------------------------------------------------------------------
// K2: attention for one (b,h), one 16-query tile. 128 thr / 4 waves; wave w
// owns keys [w*512, w*512+512). LDS: 16x2048 f32 scores (128KB, enabled by
// CDNA5's 320KB/WGP) + bf16 probs + Q/V staging (async-to-LDS).
// ---------------------------------------------------------------------------
__global__ __launch_bounds__(128) void k_attn(
    const __bf16* __restrict__ Q, const __bf16* __restrict__ K,
    const __bf16* __restrict__ V,
    float* __restrict__ attn, __bf16* __restrict__ Ob)
{
  extern __shared__ __align__(16) char smem[];
  float*  Sc = (float*)smem;                                  // 131072 B
  __bf16* Pb = (__bf16*)(smem + 131072);                      //  65536 B
  __bf16* Qt = (__bf16*)(smem + 131072 + 65536);              //   2048 B
  __bf16* Vt = (__bf16*)(smem + 131072 + 65536 + 2048);       //   4096 B

  const int tid  = threadIdx.x;
  const int lane = tid & 31;
  const int wave = tid >> 5;
  const int q0   = blockIdx.x * 16;
  const int bh   = blockIdx.y;
  const int fm   = lane & 15;
  const int kb   = (lane >> 4) * 8;

  // ---- Q tile 16x64 bf16: one async b128 per thread ----
  {
    int row = tid >> 3, col = tid & 7;
    async_b128(&Q[((size_t)bh * 2048 + q0 + row) * 64 + col * 8],
               &Qt[row * 64 + col * 8]);
  }
  wait_async0();
  __syncthreads();

  v16bf aq0 = cat8(*(const v8bf*)&Qt[fm * 64 + kb],      *(const v8bf*)&Qt[fm * 64 + 16 + kb]);
  v16bf aq1 = cat8(*(const v8bf*)&Qt[fm * 64 + 32 + kb], *(const v8bf*)&Qt[fm * 64 + 48 + kb]);

  // ---- phase 1: S = Q K^T / 8 ----
  for (int t = 0; t < 32; ++t) {
    int kt = (wave * 32 + t) * 16;
    const __bf16* krow = K + ((size_t)bh * 2048 + kt + fm) * 64;  // n=lane&15 -> key row
    v16bf bk0 = cat8(*(const v8bf*)(krow + kb),      *(const v8bf*)(krow + 16 + kb));
    v16bf bk1 = cat8(*(const v8bf*)(krow + 32 + kb), *(const v8bf*)(krow + 48 + kb));
    v8f c = {};
    c = WMMA_BF16(aq0, bk0, c);
    c = WMMA_BF16(aq1, bk1, c);
#pragma unroll
    for (int v = 0; v < 8; ++v) {
      int mm = v + 8 * (lane >> 4);
      Sc[mm * 2048 + kt + fm] = c[v] * 0.125f;   // 1/sqrt(64)
    }
  }
  __syncthreads();

  // ---- phase 2: softmax, 8 threads/row, shfl_xor reductions ----
  {
    int row = tid >> 3, sub = tid & 7;
    float mx = -3.4e38f;
    for (int j = sub; j < 2048; j += 8) mx = fmaxf(mx, Sc[row * 2048 + j]);
    for (int o = 1; o < 8; o <<= 1) mx = fmaxf(mx, __shfl_xor(mx, o));
    float sum = 0.f;
    for (int j = sub; j < 2048; j += 8) {
      float e = __expf(Sc[row * 2048 + j] - mx);
      Sc[row * 2048 + j] = e;
      sum += e;
    }
    for (int o = 1; o < 8; o <<= 1) sum += __shfl_xor(sum, o);
    float inv = 1.f / sum;
    for (int j = sub; j < 2048; j += 8) Sc[row * 2048 + j] *= inv;
  }
  __syncthreads();

  // ---- phase 3: coalesced fp32 attn write (the roofline store) + bf16 pack
  {
    float4*       dst = (float4*)(attn + ((size_t)bh * 2048 + q0) * 2048);
    const float4* src = (const float4*)Sc;
    for (int j = 0; j < 64; ++j) {
      int i = tid + j * 128;
      float4 p = src[i];
      dst[i] = p;
      v4bf pb;
      pb[0] = (__bf16)p.x; pb[1] = (__bf16)p.y; pb[2] = (__bf16)p.z; pb[3] = (__bf16)p.w;
      *(v4bf*)&Pb[i * 4] = pb;
    }
  }
  __syncthreads();

  // ---- phase 4: O = P @ V ; wave w owns dk cols [w*16, w*16+16) ----
  v8f oacc = {};
  for (int kc = 0; kc < 64; ++kc) {
    {   // stage V tile 32x64 bf16: two async b128 per thread
      int u0 = tid, u1 = tid + 128;
      async_b128(&V[((size_t)bh * 2048 + kc * 32 + (u0 >> 3)) * 64 + (u0 & 7) * 8],
                 &Vt[(u0 >> 3) * 64 + (u0 & 7) * 8]);
      async_b128(&V[((size_t)bh * 2048 + kc * 32 + (u1 >> 3)) * 64 + (u1 & 7) * 8],
                 &Vt[(u1 >> 3) * 64 + (u1 & 7) * 8]);
    }
    wait_async0();
    __syncthreads();
    v16bf ap = cat8(*(const v8bf*)&Pb[fm * 2048 + kc * 32 + kb],
                    *(const v8bf*)&Pb[fm * 2048 + kc * 32 + 16 + kb]);
    v16bf bv;
#pragma unroll
    for (int g = 0; g < 2; ++g)
#pragma unroll
      for (int w = 0; w < 8; ++w)
        bv[g * 8 + w] = Vt[(g * 16 + kb + w) * 64 + wave * 16 + fm];
    oacc = WMMA_BF16(ap, bv, oacc);
    __syncthreads();
  }

#pragma unroll
  for (int v = 0; v < 8; ++v) {
    int mm  = v + 8 * (lane >> 4);
    int s   = q0 + mm;
    int bb  = bh >> 4, h = bh & 15;
    int col = h * 64 + wave * 16 + fm;
    Ob[((size_t)bb * 2048 + s) * 1024 + col] = (__bf16)oacc[v];
  }
}

// ---------------------------------------------------------------------------
// Host-side orchestration. Workspace (bf16 elems): xb 4M | 4 weights 1M each |
// q/k/v 4M each | ob 4M  -> 48 MB.
// ---------------------------------------------------------------------------
extern "C" void kernel_launch(void* const* d_in, const int* in_sizes, int n_in,
                              void* d_out, int out_size, void* d_ws, size_t ws_size,
                              hipStream_t stream) {
  const float* x   = (const float*)d_in[0];
  const float* wq  = (const float*)d_in[1];
  const float* bq  = (const float*)d_in[2];
  const float* wk  = (const float*)d_in[3];
  const float* bk  = (const float*)d_in[4];
  const float* wv  = (const float*)d_in[5];
  const float* bv_ = (const float*)d_in[6];
  const float* wo  = (const float*)d_in[7];
  const float* bo  = (const float*)d_in[8];

  float* out  = (float*)d_out;                       // [2,2048,1024]
  float* attn = out + (size_t)2 * 2048 * 1024;       // [2,16,2048,2048]

  __bf16* ws  = (__bf16*)d_ws;
  __bf16* xb  = ws;                  // 4,194,304
  __bf16* wqb = ws + 4194304;
  __bf16* wkb = ws + 5242880;
  __bf16* wvb = ws + 6291456;
  __bf16* wob = ws + 7340032;
  __bf16* qb  = ws + 8388608;        // [bh=32][2048][64]
  __bf16* kb2 = ws + 12582912;
  __bf16* vb2 = ws + 16777216;
  __bf16* ob  = ws + 20971520;       // [4096][1024]

  k_cvt_bf16<<<4096, 256, 0, stream>>>(x,  xb,  1048576);
  k_cvt_bf16<<<1024, 256, 0, stream>>>(wq, wqb,  262144);
  k_cvt_bf16<<<1024, 256, 0, stream>>>(wk, wkb,  262144);
  k_cvt_bf16<<<1024, 256, 0, stream>>>(wv, wvb,  262144);
  k_cvt_bf16<<<1024, 256, 0, stream>>>(wo, wob,  262144);

  // QKV projections (z selects matrix); 32x64 block tiles
  k_proj<<<dim3(128, 16, 3), 128, 0, stream>>>(
      xb, wqb, wkb, wvb, bq, bk, bv_, qb, kb2, vb2, nullptr, 0);

  // attention (dynamic LDS 202752 B, within CDNA5's 320 KB/WGP)
  size_t smem = 131072 + 65536 + 2048 + 4096;
  k_attn<<<dim3(128, 32), 128, smem, stream>>>(qb, kb2, vb2, attn, ob);

  // output projection (fp32 epilogue straight into d_out)
  k_proj<<<dim3(128, 16, 1), 128, 0, stream>>>(
      ob, wob, wob, wob, bo, bo, bo, nullptr, nullptr, nullptr, out, 1);
}
